// DistMultDecoder_13907104104792
// MI455X (gfx1250) — matrix operand order repaired and checked by reading, not compiled, not run on hardware
//
#include <hip/hip_runtime.h>
#include <hip/hip_bf16.h>

typedef __attribute__((ext_vector_type(16))) _Float16 v16h;
typedef __attribute__((ext_vector_type(8)))  _Float16 v8h;
typedef __attribute__((ext_vector_type(8)))  float    v8f;
typedef __attribute__((ext_vector_type(4)))  float    v4f;

#define TILE   128
#define D_DIM  128

// LDS layout: 16-byte chunks, chunk-major: unit index = chunk*128 + row.
// XOR chunk[3:0] into the bank-relevant bits (byte addr [7:4]) so that both
// the staging stores (varying chunk, fixed row) and the fragment loads
// (varying row, fixed chunk) hit the minimal bank pattern.
// Returned offset is in _Float16 units (16B-alignment preserved).
__device__ __forceinline__ int lds_off(int chunk, int row) {
    return ((((chunk << 7) + row) << 3)) ^ ((chunk & 15) << 3);
}

__device__ __forceinline__ float sigmoidf(float x) {
    return 1.0f / (1.0f + __expf(-x));
}

__global__ void __launch_bounds__(256)
distmult_sigmoid_kernel(const float* __restrict__ xi,
                        const float* __restrict__ xj,
                        const float* __restrict__ rel,
                        float* __restrict__ out,
                        int Ni, int Nj) {
    __shared__ _Float16 ldsA[TILE * D_DIM];   // 32 KB
    __shared__ _Float16 ldsB[TILE * D_DIM];   // 32 KB

    const int t       = threadIdx.x;
    const int k       = blockIdx.z;
    const int rowBase = blockIdx.y * TILE;    // N_i tile
    const int colBase = blockIdx.x * TILE;    // N_j tile

    // ---------------- stage: f32 -> f16 into LDS (A scaled by rel[k]) --------
    {
        const int chunk = t & 15;             // which 8-element K chunk
        const int rsub  = t >> 4;             // row within 16-row pass
        const v4f r0 = *(const v4f*)(rel + (size_t)k * D_DIM + chunk * 8);
        const v4f r1 = *(const v4f*)(rel + (size_t)k * D_DIM + chunk * 8 + 4);

        #pragma unroll
        for (int p = 0; p < 8; ++p) {
            const int row = p * 16 + rsub;

            const float* arow = xi + (size_t)(rowBase + row) * D_DIM + chunk * 8;
            v4f a0 = *(const v4f*)(arow);
            v4f a1 = *(const v4f*)(arow + 4);
            v8h ah;
            ah[0] = (_Float16)(a0[0] * r0[0]);
            ah[1] = (_Float16)(a0[1] * r0[1]);
            ah[2] = (_Float16)(a0[2] * r0[2]);
            ah[3] = (_Float16)(a0[3] * r0[3]);
            ah[4] = (_Float16)(a1[0] * r1[0]);
            ah[5] = (_Float16)(a1[1] * r1[1]);
            ah[6] = (_Float16)(a1[2] * r1[2]);
            ah[7] = (_Float16)(a1[3] * r1[3]);
            *(v8h*)(ldsA + lds_off(chunk, row)) = ah;

            const float* brow = xj + (size_t)(colBase + row) * D_DIM + chunk * 8;
            v4f b0 = *(const v4f*)(brow);
            v4f b1 = *(const v4f*)(brow + 4);
            v8h bh;
            bh[0] = (_Float16)b0[0];
            bh[1] = (_Float16)b0[1];
            bh[2] = (_Float16)b0[2];
            bh[3] = (_Float16)b0[3];
            bh[4] = (_Float16)b1[0];
            bh[5] = (_Float16)b1[1];
            bh[6] = (_Float16)b1[2];
            bh[7] = (_Float16)b1[3];
            *(v8h*)(ldsB + lds_off(chunk, row)) = bh;
        }
    }
    __syncthreads();

    // ---------------- compute: 8 waves, each owns a 32x64 subtile ------------
    const int lane   = t & 31;
    const int wid    = t >> 5;
    const int laneLo = lane & 15;
    const int laneHi = lane >> 4;
    const int waveM  = (wid & 3) * 32;    // 4 waves along M (128 rows)
    const int waveN  = (wid >> 2) * 64;   // 2 waves along N (128 cols)

    v8f acc[2][4];
    #pragma unroll
    for (int i = 0; i < 2; ++i)
        #pragma unroll
        for (int j = 0; j < 4; ++j)
            acc[i][j] = (v8f){0.f, 0.f, 0.f, 0.f, 0.f, 0.f, 0.f, 0.f};

    #pragma unroll
    for (int ks = 0; ks < 4; ++ks) {         // K = 128 in steps of 32
        v16h aF[2];
        #pragma unroll
        for (int ti = 0; ti < 2; ++ti) {
            const int row = waveM + ti * 16 + laneLo;
            const int c0  = ks * 4 + laneHi;           // A: half-lane K split 8
            const v8h lo = *(const v8h*)(ldsA + lds_off(c0,     row));
            const v8h hi = *(const v8h*)(ldsA + lds_off(c0 + 2, row));
            #pragma unroll
            for (int e = 0; e < 8; ++e) { aF[ti][e] = lo[e]; aF[ti][e + 8] = hi[e]; }
        }

        v16h bF[4];
        #pragma unroll
        for (int tj = 0; tj < 4; ++tj) {
            const int row = waveN + tj * 16 + laneLo;  // column of output = row of x_j
            const int c0  = ks * 4 + laneHi;           // B: half-lane K split 8
            const v8h lo = *(const v8h*)(ldsB + lds_off(c0,     row));
            const v8h hi = *(const v8h*)(ldsB + lds_off(c0 + 2, row));
            #pragma unroll
            for (int e = 0; e < 8; ++e) { bF[tj][e] = lo[e]; bF[tj][e + 8] = hi[e]; }
        }

        #pragma unroll
        for (int ti = 0; ti < 2; ++ti)
            #pragma unroll
            for (int tj = 0; tj < 4; ++tj)
                acc[ti][tj] = __builtin_amdgcn_wmma_f32_16x16x32_f16(
                    false, aF[ti], false, bF[tj],
                    (short)0, acc[ti][tj], false, false);
    }

    // ---------------- epilogue: fused sigmoid + store ------------------------
    float* outk = out + (size_t)k * (size_t)Ni * (size_t)Nj;
    #pragma unroll
    for (int ti = 0; ti < 2; ++ti) {
        #pragma unroll
        for (int tj = 0; tj < 4; ++tj) {
            const int n = colBase + waveN + tj * 16 + laneLo;
            #pragma unroll
            for (int v = 0; v < 8; ++v) {
                const int m = rowBase + waveM + ti * 16 + laneHi * 8 + v;
                outk[(size_t)m * Nj + n] = sigmoidf(acc[ti][tj][v]);
            }
        }
    }
}

extern "C" void kernel_launch(void* const* d_in, const int* in_sizes, int n_in,
                              void* d_out, int out_size, void* d_ws, size_t ws_size,
                              hipStream_t stream) {
    const float* xi  = (const float*)d_in[0];
    const float* xj  = (const float*)d_in[1];
    const float* rel = (const float*)d_in[2];
    float* out = (float*)d_out;

    const int D  = 128;
    const int Ni = in_sizes[0] / D;   // 4096
    const int Nj = in_sizes[1] / D;   // 4096
    const int K  = in_sizes[2] / D;   // 8

    dim3 grid(Nj / TILE, Ni / TILE, K);
    dim3 block(256);
    hipLaunchKernelGGL(distmult_sigmoid_kernel, grid, block, 0, stream,
                       xi, xj, rel, out, Ni, Nj);
}
